// SelfAttentionFusion_29188597743646
// MI455X (gfx1250) — matrix operand order repaired and checked by reading, not compiled, not run on hardware
//
#include <hip/hip_runtime.h>
#include <hip/hip_bf16.h>

// ---------------------------------------------------------------------------
// SelfAttentionFusion for MI455X (gfx1250, wave32, WMMA)
//   B=8192, N=4, D=1024, H=512
// Kernel 1: QKV = X @ [Wq|Wk|Wv] + bias (M=32768, K=1024, N=2048) via
//           v_wmma_f32_16x16x32_bf16, fp32 accumulate. Double-buffered LDS,
//           B staged as k-pair-packed dwords with b128 LDS stores.
//           Q/K stored bf16 (softmax-smoothed), V stored fp32 (feeds output).
// Kernel 2: per-batch 4x4 scores, softmax over QUERY axis (dim=1), attn @ V.
// Workspace: [0, 64MiB)   QK  bf16 [32768][1024]
//            [64MiB, ...) V   fp32 [32768][1024]
// ---------------------------------------------------------------------------

typedef __attribute__((ext_vector_type(16))) __bf16 v16bf;
typedef __attribute__((ext_vector_type(8)))  float  v8f;

union FragBF16 {
    unsigned int u[8];
    v16bf        v;
};

__device__ __forceinline__ unsigned short f32_to_bf16_rne(float f) {
    unsigned int u = __float_as_uint(f);
    u += 0x7FFFu + ((u >> 16) & 1u);   // round-to-nearest-even
    return (unsigned short)(u >> 16);
}

__device__ __forceinline__ unsigned int pack_bf16x2(float lo, float hi) {
    return (unsigned int)f32_to_bf16_rne(lo) |
           ((unsigned int)f32_to_bf16_rne(hi) << 16);
}

__device__ __forceinline__ void bf16x2_to_f32(unsigned int p, float& lo, float& hi) {
    lo = __uint_as_float(p << 16);
    hi = __uint_as_float(p & 0xFFFF0000u);
}

// ---------------------------------------------------------------------------
// Kernel 1: merged QKV GEMM.
//   Block tile: BM=64 x BN=256, BK=32. 256 threads = 8 waves in 2(M) x 4(N).
//   Wave tile: 32 x 64 -> 2x4 WMMA 16x16 accumulators. Ping-pong LDS.
//   grid = (N/256, M/64) = (8, 512)
// ---------------------------------------------------------------------------
#define GEMM_BM 64
#define GEMM_BN 256
#define GEMM_BK 32
#define LDA_STRIDE 18   // dwords per A row (16 used + 2 pad; 18m mod 64 distinct
                        // for m=0..15 -> conflict-free, and even -> b64 stores)

__global__ __launch_bounds__(256)
void qkv_gemm_kernel(const float* __restrict__ f0, const float* __restrict__ f1,
                     const float* __restrict__ f2, const float* __restrict__ f3,
                     const float* __restrict__ Wq, const float* __restrict__ bq,
                     const float* __restrict__ Wk, const float* __restrict__ bk,
                     const float* __restrict__ Wv, const float* __restrict__ bv,
                     unsigned short* __restrict__ qk_out,   // bf16 [32768][1024]
                     float* __restrict__ v_out) {           // fp32 [32768][1024]
    // Ping-pong tiles: A 64x32 bf16 (k-pair packed dwords), B 16 k-pairs x 256 n
    __shared__ unsigned int lds_a[2][GEMM_BM * LDA_STRIDE];
    __shared__ unsigned int lds_b[2][(GEMM_BK / 2) * GEMM_BN];

    const int tid    = threadIdx.x;
    const int lane   = tid & 31;
    const int wid    = tid >> 5;
    const int wave_m = wid >> 2;     // 0..1
    const int wave_n = wid & 3;      // 0..3
    const int lm     = lane & 15;    // lane-in-half
    const int lh     = lane >> 4;    // half index (0: lanes 0-15, 1: lanes 16-31)

    const int n0 = blockIdx.x * GEMM_BN;   // 0..1792, multiple of 256
    const int m0 = blockIdx.y * GEMM_BM;   // row tile base in [0, 32768)

    // Select weight/bias segment for this N tile (256-wide tiles never straddle
    // the 512/1024 boundaries).
    const float* W;  const float* bias;  int ldw; int coff;
    if (n0 < 512)        { W = Wq; bias = bq; ldw = 512;  coff = n0;        }
    else if (n0 < 1024)  { W = Wk; bias = bk; ldw = 512;  coff = n0 - 512;  }
    else                 { W = Wv; bias = bv; ldw = 1024; coff = n0 - 1024; }

    const float* fsel[4] = {f0, f1, f2, f3};

    // Stage one BK-slab of A and B (fp32 -> bf16) into buffer p.
    auto fill_tiles = [&](int kt, int p) {
        const int kbase = kt * GEMM_BK;
        unsigned int* la = lds_a[p];
        unsigned int* lb = lds_b[p];
        // A: 64x32 f32 = 512 float4, 2 per thread; one b64 LDS store each
        #pragma unroll
        for (int it = 0; it < 2; ++it) {
            const int f4i = it * 256 + tid;          // 0..511
            const int m   = f4i >> 3;                // 8 float4 per row
            const int kl  = (f4i & 7) * 4;           // k offset in tile
            const int r   = m0 + m;                  // global X row
            const float* src = fsel[r & 3] + (size_t)(r >> 2) * 1024 + kbase + kl;
            const float4 e = *(const float4*)src;
            uint2 pk;
            pk.x = pack_bf16x2(e.x, e.y);
            pk.y = pack_bf16x2(e.z, e.w);
            *(uint2*)(la + m * LDA_STRIDE + (kl >> 1)) = pk;   // 8B aligned
        }
        // B: per k-pair, load both k rows at same n (2x coalesced float4),
        // pack pairs into dwords, one b128 LDS store. 1024 pair-loads, 4/thread.
        #pragma unroll
        for (int it = 0; it < 4; ++it) {
            const int idx = it * 256 + tid;          // 0..1023
            const int kp  = idx >> 6;                // 0..15
            const int n   = (idx & 63) * 4;
            const float* src0 = W + (size_t)(kbase + 2 * kp) * ldw + coff + n;
            const float4 e0 = *(const float4*)src0;
            const float4 e1 = *(const float4*)(src0 + ldw);
            uint4 pk;
            pk.x = pack_bf16x2(e0.x, e1.x);
            pk.y = pack_bf16x2(e0.y, e1.y);
            pk.z = pack_bf16x2(e0.z, e1.z);
            pk.w = pack_bf16x2(e0.w, e1.w);
            *(uint4*)(lb + kp * GEMM_BN + n) = pk;             // 16B aligned
        }
    };

    v8f acc[2][4] = {};

    fill_tiles(0, 0);

    const int KSTEPS = 1024 / GEMM_BK;   // 32
    for (int kt = 0; kt < KSTEPS; ++kt) {
        __syncthreads();   // fill(kt) visible; buf[(kt+1)&1] readers (kt-1) done
        if (kt + 1 < KSTEPS) fill_tiles(kt + 1, (kt + 1) & 1);

        const unsigned int* la = lds_a[kt & 1];
        const unsigned int* lb = lds_b[kt & 1];

        // Fragments per documented 16-bit layouts (ISA 7.12.2)
        FragBF16 fa[2];
        #pragma unroll
        for (int mt = 0; mt < 2; ++mt) {
            const int mrow = wave_m * 32 + mt * 16 + lm;
            #pragma unroll
            for (int r = 0; r < 8; ++r) {
                // A: VGPR r holds k-pair index (r/4)*8 + half*4 + (r%4)
                const int kp = ((r >> 2) * 8) + (lh * 4) + (r & 3);
                fa[mt].u[r] = la[mrow * LDA_STRIDE + kp];
            }
        }
        FragBF16 fb[4];
        #pragma unroll
        for (int nt = 0; nt < 4; ++nt) {
            const int ncol = wave_n * 64 + nt * 16 + lm;
            #pragma unroll
            for (int r = 0; r < 8; ++r) {
                // B: VGPR r holds k-pair index (r/4)*8 + half*4 + (r%4) for column ncol
                fb[nt].u[r] = lb[(((r >> 2) * 8) + (lh * 4) + (r & 3)) * GEMM_BN + ncol];
            }
        }

        // 8 WMMAs per K-step, A fragment reused across N subtiles
        #pragma unroll
        for (int mt = 0; mt < 2; ++mt)
            #pragma unroll
            for (int nt = 0; nt < 4; ++nt)
                acc[mt][nt] = __builtin_amdgcn_wmma_f32_16x16x32_bf16(
                    false, fa[mt].v, false, fb[nt].v,
                    (short)0, acc[mt][nt], false, false);
    }

    // ---- Epilogue: add bias; Q/K -> bf16 buffer, V -> fp32 buffer ----
    #pragma unroll
    for (int mt = 0; mt < 2; ++mt) {
        #pragma unroll
        for (int nt = 0; nt < 4; ++nt) {
            const int col_l = wave_n * 64 + nt * 16 + lm;     // 0..255 in tile
            const float bb = bias[coff + col_l];
            #pragma unroll
            for (int r = 0; r < 8; ++r) {
                // C/D layout: VGPR r, lane -> row = r + 8*half, col = lane%16
                const int   row = m0 + wave_m * 32 + mt * 16 + lh * 8 + r;
                const float val = acc[mt][nt][r] + bb;
                if (n0 < 1024) {
                    // QK buffer: col 0..511 = q, 512..1023 = k
                    const size_t cg = (size_t)(n0 + col_l);
                    qk_out[(size_t)row * 1024 + cg] = f32_to_bf16_rne(val);
                } else {
                    const size_t cg = (size_t)(n0 - 1024 + col_l);
                    v_out[(size_t)row * 1024 + cg] = val;
                }
            }
        }
    }
}

// ---------------------------------------------------------------------------
// Kernel 2: per-batch attention. One wave per batch element.
//   QK rows (bf16, stride 1024): row = 4*b + n ; cols [0,512)=q, [512,1024)=k
//   V rows (fp32, stride 1024).
//   softmax over QUERY axis (per-column over n), then attn @ V.
// ---------------------------------------------------------------------------
__global__ __launch_bounds__(256)
void attn_kernel(const unsigned short* __restrict__ qk,
                 const float* __restrict__ vbuf,
                 float* __restrict__ out) {
    const int wave = blockIdx.x * (blockDim.x >> 5) + (threadIdx.x >> 5);
    const int lane = threadIdx.x & 31;
    const size_t row0 = (size_t)wave * 4;

    // scores[n][m] = dot(q[n], k[m]) over H=512; lanes split H, 4 elems/pass
    float s[16] = {};
    #pragma unroll
    for (int h0 = 0; h0 < 512; h0 += 128) {
        const int h = h0 + lane * 4;
        float qf[4][4], kf[4][4];
        #pragma unroll
        for (int n = 0; n < 4; ++n) {
            const uint2 qp = *(const uint2*)(qk + (row0 + n) * 1024 + h);
            const uint2 kp = *(const uint2*)(qk + (row0 + n) * 1024 + 512 + h);
            bf16x2_to_f32(qp.x, qf[n][0], qf[n][1]);
            bf16x2_to_f32(qp.y, qf[n][2], qf[n][3]);
            bf16x2_to_f32(kp.x, kf[n][0], kf[n][1]);
            bf16x2_to_f32(kp.y, kf[n][2], kf[n][3]);
        }
        #pragma unroll
        for (int n = 0; n < 4; ++n)
            #pragma unroll
            for (int m = 0; m < 4; ++m)
                s[n * 4 + m] += qf[n][0] * kf[m][0] + qf[n][1] * kf[m][1] +
                                qf[n][2] * kf[m][2] + qf[n][3] * kf[m][3];
    }
    // wave-wide reduction; afterwards every lane holds the full 4x4 scores
    #pragma unroll
    for (int i = 0; i < 16; ++i) {
        float v = s[i];
        #pragma unroll
        for (int off = 16; off >= 1; off >>= 1)
            v += __shfl_xor(v, off, 32);
        s[i] = v;
    }

    // softmax over the query axis n, independently per column m (dim=1)
    float attn[16];
    #pragma unroll
    for (int m = 0; m < 4; ++m) {
        const float mx = fmaxf(fmaxf(s[m], s[4 + m]), fmaxf(s[8 + m], s[12 + m]));
        const float e0 = expf(s[m]      - mx);
        const float e1 = expf(s[4 + m]  - mx);
        const float e2 = expf(s[8 + m]  - mx);
        const float e3 = expf(s[12 + m] - mx);
        const float inv = 1.0f / (e0 + e1 + e2 + e3);
        attn[m]      = e0 * inv;
        attn[4 + m]  = e1 * inv;
        attn[8 + m]  = e2 * inv;
        attn[12 + m] = e3 * inv;
    }

    // attended[n][d] = sum_m attn[n][m] * v[m][d] ; lanes split D=1024
    const float* vp = vbuf + row0 * 1024;
    float* ob = out + (size_t)wave * 4096;
    #pragma unroll
    for (int d0 = 0; d0 < 1024; d0 += 128) {
        const int d = d0 + lane * 4;
        float4 vv[4];
        #pragma unroll
        for (int m = 0; m < 4; ++m)
            vv[m] = *(const float4*)(vp + m * 1024 + d);
        #pragma unroll
        for (int n = 0; n < 4; ++n) {
            float4 o;
            o.x = attn[n*4+0]*vv[0].x + attn[n*4+1]*vv[1].x + attn[n*4+2]*vv[2].x + attn[n*4+3]*vv[3].x;
            o.y = attn[n*4+0]*vv[0].y + attn[n*4+1]*vv[1].y + attn[n*4+2]*vv[2].y + attn[n*4+3]*vv[3].y;
            o.z = attn[n*4+0]*vv[0].z + attn[n*4+1]*vv[1].z + attn[n*4+2]*vv[2].z + attn[n*4+3]*vv[3].z;
            o.w = attn[n*4+0]*vv[0].w + attn[n*4+1]*vv[1].w + attn[n*4+2]*vv[2].w + attn[n*4+3]*vv[3].w;
            *(float4*)(ob + n * 1024 + d) = o;
        }
    }
}

// ---------------------------------------------------------------------------
extern "C" void kernel_launch(void* const* d_in, const int* in_sizes, int n_in,
                              void* d_out, int out_size, void* d_ws, size_t ws_size,
                              hipStream_t stream) {
    (void)in_sizes; (void)n_in; (void)out_size; (void)ws_size;
    const float* f0 = (const float*)d_in[0];
    const float* f1 = (const float*)d_in[1];
    const float* f2 = (const float*)d_in[2];
    const float* f3 = (const float*)d_in[3];
    const float* Wq = (const float*)d_in[4];
    const float* bq = (const float*)d_in[5];
    const float* Wk = (const float*)d_in[6];
    const float* bk = (const float*)d_in[7];
    const float* Wv = (const float*)d_in[8];
    const float* bv = (const float*)d_in[9];

    // Workspace layout: QK bf16 (64 MiB) then V fp32 (128 MiB)
    unsigned short* qk = (unsigned short*)d_ws;
    float* vbuf = (float*)((char*)d_ws + ((size_t)32768 * 1024 * 2));

    // GEMM: grid (N/256, M/64) = (8, 512)
    dim3 g1(2048 / GEMM_BN, 32768 / GEMM_BM);
    qkv_gemm_kernel<<<g1, 256, 0, stream>>>(f0, f1, f2, f3,
                                            Wq, bq, Wk, bk, Wv, bv, qk, vbuf);

    // Attention: 8192 batches, 1 wave each, 8 waves per 256-thread block
    attn_kernel<<<8192 / 8, 256, 0, stream>>>(qk, vbuf, (float*)d_out);
}
